// VQ_VAE_MultiCells_17136919511060
// MI455X (gfx1250) — matrix operand order repaired and checked by reading, not compiled, not run on hardware
//
#include <hip/hip_runtime.h>
#include <math.h>

typedef __bf16 bf16;
typedef __bf16 v16bf __attribute__((ext_vector_type(16)));
typedef __bf16 v8bf  __attribute__((ext_vector_type(8)));
typedef float  v8f   __attribute__((ext_vector_type(8)));
typedef __attribute__((address_space(3))) bf16 lds_bf16;

#define B_  4096
#define C_  8
#define G_  978
#define D0_ 512
#define D1_ 256
#define D2_ 128
#define K_  512

// GEMM tiling: 128(M) x 64(N) tile, K-step 32, 256 threads = 8 waves (wave32)
#define TM 128
#define TN 64
#define TK 32
#define APAD 8
#define BPAD 8

// Generic C = tanh(A @ W + bias) bf16-WMMA GEMM.
// CONTRACT: M % TM == 0 (all call sites use M = 4096).
// ASRC = 0: A is fp32 (Af).  ASRC = 1: A is bf16 (Ab), staged via async-to-LDS.
// A: M x Kdim, row stride lda, per-cell offset aCell.
// W: Kdim x N fp32 row-major (ldw), per-cell offset wCell.
// Outputs optionally fp32 (outF/ldoF) and/or bf16 (outB/ldoB).
template <int ASRC>
__global__ __launch_bounds__(256)
void gemm_bf16_wmma(const float* __restrict__ Af, const bf16* __restrict__ Ab,
                    long long aCell, int lda,
                    const float* __restrict__ W, long long wCell, int ldw,
                    const float* __restrict__ bias, long long bCell,
                    float* __restrict__ outF, long long ofCell, int ldoF,
                    bf16* __restrict__ outB, long long obCell, int ldoB,
                    int M, int N, int Kdim, int doTanh)
{
    __shared__ bf16 As[TM][TK + APAD];          // A tile, row-major (M x K)
    __shared__ bf16 Bt[TN][TK + BPAD];          // B tile, TRANSPOSED (N x K)

    const int tid   = threadIdx.x;
    const int lane  = tid & 31;
    const int wave  = tid >> 5;     // 0..7
    const int waveM = wave & 3;     // 4 waves over M (32 rows each)
    const int waveN = wave >> 2;    // 2 waves over N (32 cols each)
    const int lmod  = lane & 15;
    const int lhalf = lane >> 4;

    const int tileN = blockIdx.x * TN;
    const int tileM = blockIdx.y * TM;
    const int cell  = blockIdx.z;

    const float* Afp = (ASRC == 0) ? (Af + (long long)cell * aCell) : nullptr;
    const bf16*  Abp = (ASRC == 1) ? (Ab + (long long)cell * aCell) : nullptr;
    const float* Wp  = W + (long long)cell * wCell;
    const float* bp  = bias ? (bias + (long long)cell * bCell) : nullptr;

    v8f acc[2][2];
    #pragma unroll
    for (int i = 0; i < 2; ++i)
        #pragma unroll
        for (int j = 0; j < 2; ++j)
            #pragma unroll
            for (int e = 0; e < 8; ++e) acc[i][j][e] = 0.0f;

    // cooperative tile-load assignments
    const int aRow     = tid >> 1;         // 0..127
    const int aColBase = (tid & 1) * 16;   // 0 / 16
    const int bRow     = tid >> 3;         // 0..31  (K index)
    const int bColBase = (tid & 7) * 8;    // 0..56  (N index)

    const bool nFull = (tileN + TN) <= N;
    const int  gmA   = tileM + aRow;       // always < M (contract)

    // per-row base pointers (k varies in the loop)
    const float* aRowF = (ASRC == 0) ? (Afp + (long long)gmA * lda) : nullptr;
    const bf16*  aRowB = (ASRC == 1) ? (Abp + (long long)gmA * lda) : nullptr;

    // this thread's LDS destination for the A tile (32 bytes, 16B aligned)
    lds_bf16* aDst = (lds_bf16*)&As[aRow][aColBase];

    // fragment-load + 4x WMMA step (shared by main loop and K-tail)
    auto mma_step = [&]() {
        v16bf afrag[2], bfrag[2];
        #pragma unroll
        for (int mi = 0; mi < 2; ++mi) {
            const int row = waveM * 32 + mi * 16 + lmod;
            const int kb  = lhalf * 8;
            const v8bf a0 = *(const v8bf*)&As[row][kb];
            const v8bf a1 = *(const v8bf*)&As[row][16 + kb];
            #pragma unroll
            for (int i = 0; i < 8; ++i) {
                afrag[mi][i]     = a0[i];
                afrag[mi][8 + i] = a1[i];
            }
        }
        #pragma unroll
        for (int ni = 0; ni < 2; ++ni) {
            const int col = waveN * 32 + ni * 16 + lmod;
            const v8bf b0 = *(const v8bf*)&Bt[col][lhalf * 8];
            const v8bf b1 = *(const v8bf*)&Bt[col][16 + lhalf * 8];
            #pragma unroll
            for (int i = 0; i < 8; ++i) {
                bfrag[ni][i]     = b0[i];
                bfrag[ni][8 + i] = b1[i];
            }
        }
        #pragma unroll
        for (int mi = 0; mi < 2; ++mi)
            #pragma unroll
            for (int ni = 0; ni < 2; ++ni)
                acc[mi][ni] = __builtin_amdgcn_wmma_f32_16x16x32_bf16(
                    false, afrag[mi], false, bfrag[ni],
                    (short)0, acc[mi][ni], false, false);
    };

    const int kMain = Kdim & ~(TK - 1);

    // ---------------- main K loop: no bounds checks at all ----------------
    for (int k0 = 0; k0 < kMain; k0 += TK) {
        // A tile (128x32) -> bf16 LDS
        if (ASRC == 0) {
            // fp32 source: load + convert through VGPRs
            const float2* p = (const float2*)(aRowF + k0 + aColBase); // 8B aligned
            bf16 tmp[16];
            #pragma unroll
            for (int i = 0; i < 8; ++i) {
                const float2 f = p[i];
                tmp[2 * i]     = (bf16)f.x;
                tmp[2 * i + 1] = (bf16)f.y;
            }
            *(uint4*)&As[aRow][aColBase]     = *(const uint4*)&tmp[0];
            *(uint4*)&As[aRow][aColBase + 8] = *(const uint4*)&tmp[8];
        } else {
            // bf16 source: pure copy -> async global->LDS, bypass VGPRs.
            // INST_OFFSET applies to both LDS and global address (ISA 15.18.3),
            // so one base + offset:16 covers this thread's 32 bytes.
            const bf16* gsrc = aRowB + k0 + aColBase;                 // 16B aligned
            asm volatile("global_load_async_to_lds_b128 %0, %1, off"
                         :: "v"(aDst), "v"(gsrc) : "memory");
            asm volatile("global_load_async_to_lds_b128 %0, %1, off offset:16"
                         :: "v"(aDst), "v"(gsrc) : "memory");
        }

        // B tile (32x64) -> bf16 LDS, stored transposed Bt[n][k]
        {
            const int gk = k0 + bRow;                                 // < Kdim
            const float* wRow = Wp + (long long)gk * ldw + tileN;
            if (nFull) {
                const float2* p = (const float2*)(wRow + bColBase);   // 8B aligned
                #pragma unroll
                for (int i = 0; i < 4; ++i) {
                    const float2 f = p[i];
                    Bt[bColBase + 2 * i][bRow]     = (bf16)f.x;
                    Bt[bColBase + 2 * i + 1][bRow] = (bf16)f.y;
                }
            } else {
                // branchless clamped loads (N-tail tiles only)
                #pragma clang loop unroll(disable)
                for (int i = 0; i < 8; ++i) {
                    const int gn  = tileN + bColBase + i;
                    const int gnc = (gn < N) ? gn : (N - 1);
                    const float v = Wp[(long long)gk * ldw + gnc];
                    Bt[bColBase + i][bRow] = (gn < N) ? (bf16)v : (bf16)0.0f;
                }
            }
        }

        // prefetch next K-tile (speculative; dropped if OOB)
        if (ASRC == 0) __builtin_prefetch(aRowF + k0 + TK + aColBase, 0, 1);
        else           __builtin_prefetch(aRowB + k0 + TK + aColBase, 0, 1);
        __builtin_prefetch(Wp + (long long)(k0 + TK + bRow) * ldw + tileN + bColBase, 0, 1);

        // drain this thread's async copies; barrier publishes whole tile
        if (ASRC == 1)
            asm volatile("s_wait_asynccnt 0x0" ::: "memory");
        __syncthreads();
        mma_step();
        __syncthreads();
    }

    // ---------------- single K-tail block (Kdim % 32 != 0) ----------------
    if (kMain < Kdim) {
        #pragma clang loop unroll(disable)
        for (int i = 0; i < 16; ++i) {
            const int gk = kMain + aColBase + i;
            bf16 v = (bf16)0.0f;
            if (gk < Kdim) {
                if (ASRC == 0) v = (bf16)aRowF[gk];
                else           v = aRowB[gk];
            }
            As[aRow][aColBase + i] = v;
        }
        {
            const int gk = kMain + bRow;
            #pragma clang loop unroll(disable)
            for (int i = 0; i < 8; ++i) {
                const int gn = tileN + bColBase + i;
                bf16 v = (bf16)0.0f;
                if (gk < Kdim && gn < N)
                    v = (bf16)Wp[(long long)gk * ldw + gn];
                Bt[bColBase + i][bRow] = v;
            }
        }
        __syncthreads();
        mma_step();
    }

    // ---- epilogue: bias + tanh, write fp32 / bf16 ----
    float* outFp = outF ? (outF + (long long)cell * ofCell) : nullptr;
    bf16*  outBp = outB ? (outB + (long long)cell * obCell) : nullptr;

    #pragma unroll
    for (int mi = 0; mi < 2; ++mi) {
        #pragma unroll
        for (int ni = 0; ni < 2; ++ni) {
            const int gn = tileN + waveN * 32 + ni * 16 + lmod;
            if (gn >= N) continue;
            const float bv = bp ? bp[gn] : 0.0f;
            #pragma unroll
            for (int r = 0; r < 8; ++r) {
                const int gm = tileM + waveM * 32 + mi * 16 + lhalf * 8 + r;
                float v = acc[mi][ni][r] + bv;
                if (doTanh) v = tanhf(v);
                if (outFp) outFp[(long long)gm * ldoF + gn] = v;
                if (outBp) outBp[(long long)gm * ldoB + gn] = (bf16)v;
            }
        }
    }
}

// VQ: one block per batch row. argmin over K_ codes (D2_ dims), gather q,
// write q_st (fp32 out) + q (bf16 for decoder), accumulate sum((q-z)^2).
__global__ __launch_bounds__(256)
void vq_kernel(const float* __restrict__ z, const float* __restrict__ cb,
               bf16* __restrict__ qbf, float* __restrict__ qst,
               float* __restrict__ lossAcc)
{
    __shared__ float zs[D2_];
    __shared__ float dists[256];
    __shared__ int   idxs[256];
    __shared__ int   bestIdx;
    __shared__ float partial[256];

    const int b = blockIdx.x;
    const int t = threadIdx.x;
    if (t < D2_) zs[t] = z[(long long)b * D2_ + t];
    __syncthreads();

    float best = 3.4e38f; int bi = 0x7fffffff;
    for (int e = t; e < K_; e += 256) {
        const float4* crow = (const float4*)(cb + (long long)e * D2_);
        float d = 0.0f;
        #pragma unroll 8
        for (int j = 0; j < D2_ / 4; ++j) {
            const float4 c4 = crow[j];
            const float4 z4 = *(const float4*)&zs[4 * j];
            float df;
            df = c4.x - z4.x; d += df * df;
            df = c4.y - z4.y; d += df * df;
            df = c4.z - z4.z; d += df * df;
            df = c4.w - z4.w; d += df * df;
        }
        if (d < best) { best = d; bi = e; }
    }
    dists[t] = best; idxs[t] = bi;
    __syncthreads();
    for (int s = 128; s > 0; s >>= 1) {
        if (t < s) {
            const float od = dists[t + s]; const int oi = idxs[t + s];
            if (od < dists[t] || (od == dists[t] && oi < idxs[t])) {
                dists[t] = od; idxs[t] = oi;
            }
        }
        __syncthreads();
    }
    if (t == 0) bestIdx = idxs[0];
    __syncthreads();

    const int e = bestIdx;
    float p = 0.0f;
    if (t < D2_) {
        const float qv = cb[(long long)e * D2_ + t];
        qbf[(long long)b * D2_ + t] = (bf16)qv;
        qst[(long long)b * D2_ + t] = qv;         // q_st == q numerically
        const float df = qv - zs[t];
        p = df * df;
    }
    partial[t] = p;
    __syncthreads();
    for (int s = 128; s > 0; s >>= 1) {
        if (t < s) partial[t] += partial[t + s];
        __syncthreads();
    }
    if (t == 0) atomicAdd(lossAcc, partial[0]);
}

__global__ void zero_kernel(float* p) {
    if (threadIdx.x == 0 && blockIdx.x == 0) *p = 0.0f;
}

// loss = q_latent + 0.25*e_latent = 1.25 * mean((q - z)^2)
__global__ void finalize_loss(const float* __restrict__ acc, float* __restrict__ out) {
    if (threadIdx.x == 0 && blockIdx.x == 0)
        out[0] = 1.25f * (*acc) / (float)((long long)B_ * D2_);
}

extern "C" void kernel_launch(void* const* d_in, const int* in_sizes, int n_in,
                              void* d_out, int out_size, void* d_ws, size_t ws_size,
                              hipStream_t stream)
{
    const float* inputs   = (const float*)d_in[0];
    const float* enc_w    = (const float*)d_in[1];
    const float* enc_b    = (const float*)d_in[2];
    const float* w1       = (const float*)d_in[3];
    const float* b1       = (const float*)d_in[4];
    const float* w2       = (const float*)d_in[5];
    const float* b2       = (const float*)d_in[6];
    const float* codebook = (const float*)d_in[7];
    const float* dw1      = (const float*)d_in[8];
    const float* db1      = (const float*)d_in[9];
    const float* dw2      = (const float*)d_in[10];
    const float* db2      = (const float*)d_in[11];
    const float* dec_w    = (const float*)d_in[12];
    const float* dec_b    = (const float*)d_in[13];

    float* out      = (float*)d_out;
    float* loss_out = out;
    float* xrecon   = out + 1;
    float* qst      = out + 1 + (long long)B_ * C_ * G_;

    // workspace layout (~41 MB): h aliased with r2 (h dead after stage 2)
    char*  ws  = (char*)d_ws;
    size_t off = 0;
    bf16*  h   = (bf16*)(ws + off); off += (size_t)B_ * (C_ * D0_) * 2;  // 32 MB
    bf16*  z1  = (bf16*)(ws + off); off += (size_t)B_ * D1_ * 2;          //  2 MB
    float* zf  = (float*)(ws + off); off += (size_t)B_ * D2_ * 4;         //  2 MB
    bf16*  qb  = (bf16*)(ws + off); off += (size_t)B_ * D2_ * 2;          //  1 MB
    bf16*  r1  = (bf16*)(ws + off); off += (size_t)B_ * D1_ * 2;          //  2 MB
    float* lacc = (float*)(ws + off);
    bf16*  r2  = h;

    const dim3 blk(256);
    const int gy = B_ / TM;

    // 1) per-cell encoders: h[:,c,:] = tanh(inputs[:,c,:] @ enc_w[c] + enc_b[c])
    gemm_bf16_wmma<0><<<dim3((D0_ + TN - 1) / TN, gy, C_), blk, 0, stream>>>(
        inputs, nullptr, (long long)G_, C_ * G_,
        enc_w, (long long)G_ * D0_, D0_,
        enc_b, (long long)D0_,
        nullptr, 0, 0,
        h, (long long)D0_, C_ * D0_,
        B_, D0_, G_, 1);

    // 2) z1 = tanh(h @ w1 + b1)
    gemm_bf16_wmma<1><<<dim3((D1_ + TN - 1) / TN, gy, 1), blk, 0, stream>>>(
        nullptr, h, 0, C_ * D0_,
        w1, 0, D1_,
        b1, 0,
        nullptr, 0, 0,
        z1, 0, D1_,
        B_, D1_, C_ * D0_, 1);

    // 3) z = tanh(z1 @ w2 + b2)  (fp32 for VQ)
    gemm_bf16_wmma<1><<<dim3((D2_ + TN - 1) / TN, gy, 1), blk, 0, stream>>>(
        nullptr, z1, 0, D1_,
        w2, 0, D2_,
        b2, 0,
        zf, 0, D2_,
        nullptr, 0, 0,
        B_, D2_, D1_, 1);

    // 4) vector quantizer + loss
    zero_kernel<<<1, 32, 0, stream>>>(lacc);
    vq_kernel<<<B_, 256, 0, stream>>>(zf, codebook, qb, qst, lacc);
    finalize_loss<<<1, 32, 0, stream>>>(lacc, loss_out);

    // 5) r1 = tanh(q @ dw1 + db1)
    gemm_bf16_wmma<1><<<dim3((D1_ + TN - 1) / TN, gy, 1), blk, 0, stream>>>(
        nullptr, qb, 0, D2_,
        dw1, 0, D1_,
        db1, 0,
        nullptr, 0, 0,
        r1, 0, D1_,
        B_, D1_, D2_, 1);

    // 6) r2 = tanh(r1 @ dw2 + db2)   (reuses h's workspace)
    gemm_bf16_wmma<1><<<dim3((C_ * D0_ + TN - 1) / TN, gy, 1), blk, 0, stream>>>(
        nullptr, r1, 0, D1_,
        dw2, 0, C_ * D0_,
        db2, 0,
        nullptr, 0, 0,
        r2, 0, C_ * D0_,
        B_, C_ * D0_, D1_, 1);

    // 7) per-cell decoders: x_recon[:,c,:] = r2[:,c*D0:(c+1)*D0] @ dec_w[c] + dec_b[c]
    gemm_bf16_wmma<1><<<dim3((G_ + TN - 1) / TN, gy, C_), blk, 0, stream>>>(
        nullptr, r2, (long long)D0_, C_ * D0_,
        dec_w, (long long)D0_ * G_, G_,
        dec_b, (long long)G_,
        xrecon, (long long)G_, C_ * G_,
        nullptr, 0, 0,
        B_, G_, D0_, 0);
}
